// Joint_Attention_Layer_73126113182197
// MI455X (gfx1250) — matrix operand order repaired and checked by reading, not compiled, not run on hardware
//
#include <hip/hip_runtime.h>
#include <hip/hip_bf16.h>

// ---------------------------------------------------------------------------
// Joint attention, MI455X (gfx1250), wave32 + v_wmma_f32_16x16x32_bf16.
//   Q = img@Wq^T, K = text@Wk^T  (bf16, f32 accum)
//   flash-style: pass1 row stats (m,l) split over 8 key chunks for occupancy,
//   combine, pass2 O = (exp(S-m)/l * 1/16) @ [img|text]
// ---------------------------------------------------------------------------

typedef __attribute__((ext_vector_type(16))) __bf16 v16bf;
typedef __attribute__((ext_vector_type(8)))  float  v8f;

union Frag16 { v16bf v; uint4 q[2]; __bf16 e[16]; };
union Frag8  { v8f   v; float f[8]; };

#define NROW    8192
#define DIM     256
#define VW      512   // concat(img,text) width
#define KCHUNKS 8     // key-dim split for the stats pass (occupancy)
#define KCHLEN  (NROW / KCHUNKS)

// ---- fragment loaders (layouts per CDNA5 ISA 7.12.2, wave32) ---------------
// A 16x32 bf16: lanes 0-15 row m=lane, k in {0..7}U{16..23}; lanes 16-31 row
// m=lane-16, k in {8..15}U{24..31}  -> two contiguous 16B runs per lane.
__device__ inline v16bf load_a_bf16(const __bf16* base, int ld, int row0, int k0) {
    int lane = threadIdx.x & 31;
    int m    = lane & 15;
    int kb   = (lane < 16) ? 0 : 8;
    const __bf16* p = base + (size_t)(row0 + m) * ld + k0 + kb;
    Frag16 f;
    f.q[0] = *reinterpret_cast<const uint4*>(p);        // k = kb .. kb+7
    f.q[1] = *reinterpret_cast<const uint4*>(p + 16);   // k = kb+16 .. kb+23
    return f.v;
}

__device__ inline v16bf load_a_f32(const float* base, int ld, int row0, int k0) {
    int lane = threadIdx.x & 31;
    int m    = lane & 15;
    int kb   = (lane < 16) ? 0 : 8;
    const float* p = base + (size_t)(row0 + m) * ld + k0 + kb;
    Frag16 f;
#pragma unroll
    for (int j = 0; j < 8; ++j) f.e[j]     = (__bf16)p[j];
#pragma unroll
    for (int j = 0; j < 8; ++j) f.e[8 + j] = (__bf16)p[16 + j];
    return f.v;
}

// B 32x16 bf16: lane n=lane&15, k in {kb..kb+7}U{kb+16..kb+23}, kb = (lane<16 ? 0 : 8).
// Source stored "k-contiguous per n": element B[k][n] at base + (n0+n)*ld + k0+k.
__device__ inline v16bf load_b_bf16(const __bf16* base, size_t ld, int n0, size_t k0) {
    int lane = threadIdx.x & 31;
    int n    = lane & 15;
    int kb   = (lane < 16) ? 0 : 8;
    const __bf16* p = base + (size_t)(n0 + n) * ld + k0 + kb;
    Frag16 f;
    f.q[0] = *reinterpret_cast<const uint4*>(p);
    f.q[1] = *reinterpret_cast<const uint4*>(p + 16);
    return f.v;
}

__device__ inline v16bf load_b_f32(const float* base, int ld, int n0, int k0) {
    int lane = threadIdx.x & 31;
    int n    = lane & 15;
    int kb   = (lane < 16) ? 0 : 8;
    const float* p = base + (size_t)(n0 + n) * ld + k0 + kb;
    Frag16 f;
#pragma unroll
    for (int j = 0; j < 8; ++j) f.e[j] = (__bf16)p[j];
#pragma unroll
    for (int j = 0; j < 8; ++j) f.e[8 + j] = (__bf16)p[16 + j];
    return f.v;
}

__device__ inline v8f wmma_bf16(v16bf a, v16bf b, v8f c) {
    return __builtin_amdgcn_wmma_f32_16x16x32_bf16(false, a, false, b, (short)0, c,
                                                   false, false);
}

// ---- kernel 1: Out[i][j] = bf16( sum_d X[i][d] * W[j][d] ) -----------------
__global__ __launch_bounds__(128) void proj_bf16_kernel(const float* __restrict__ X,
                                                        const float* __restrict__ W,
                                                        __bf16* __restrict__ Out) {
    int wid  = threadIdx.x >> 5;
    int lane = threadIdx.x & 31;
    int i0   = blockIdx.x * 16;
    int j0   = (blockIdx.y * 4 + wid) * 16;
    v8f acc = {};
#pragma unroll
    for (int d = 0; d < 8; ++d) {
        v16bf a = load_a_f32(X, DIM, i0, d * 32);
        v16bf b = load_b_f32(W, DIM, j0, d * 32);   // B[d][n] = W[j0+n][d]
        acc = wmma_bf16(a, b, acc);
    }
    Frag8 A; A.v = acc;
    int n  = lane & 15;
    int ro = (lane < 16) ? 0 : 8;
#pragma unroll
    for (int j = 0; j < 8; ++j)
        Out[(size_t)(i0 + ro + j) * DIM + j0 + n] = (__bf16)A.f[j];
}

// ---- kernel 2: Vt[c][i] = bf16(concat(img,text)[i][c]) ---------------------
__global__ void build_vt_kernel(const float* __restrict__ img,
                                const float* __restrict__ text,
                                __bf16* __restrict__ Vt) {
    int i = blockIdx.x * 256 + threadIdx.x;
    int c = blockIdx.y;
    float v = (c < DIM) ? img[(size_t)i * DIM + c]
                        : text[(size_t)i * DIM + (c - DIM)];
    Vt[(size_t)c * NROW + i] = (__bf16)v;
}

// ---- kernel 3: partial row max & sum(exp) over a 1024-key chunk ------------
__global__ __launch_bounds__(128) void attn_stats_kernel(const __bf16* __restrict__ Q,
                                                         const __bf16* __restrict__ K,
                                                         float* __restrict__ mpart,
                                                         float* __restrict__ lpart) {
    int wid   = threadIdx.x >> 5;
    int lane  = threadIdx.x & 31;
    int q0    = (blockIdx.x * 4 + wid) * 16;
    int chunk = blockIdx.y;
    int kbeg  = chunk * KCHLEN;

    v16bf qa[8];
#pragma unroll
    for (int d = 0; d < 8; ++d) qa[d] = load_a_bf16(Q, DIM, q0, d * 32);

    float m[8], l[8];
#pragma unroll
    for (int j = 0; j < 8; ++j) { m[j] = -__builtin_inff(); l[j] = 0.f; }

    for (int k0 = kbeg; k0 < kbeg + KCHLEN; k0 += 32) {
        if (k0 + 32 < kbeg + KCHLEN)
            __builtin_prefetch(K + (size_t)(k0 + 32 + (lane & 15)) * DIM, 0, 1);
        v8f s0 = {}, s1 = {};
#pragma unroll
        for (int d = 0; d < 8; ++d) {
            v16bf b0 = load_b_bf16(K, DIM, k0,      (size_t)d * 32); // B[d][key]
            v16bf b1 = load_b_bf16(K, DIM, k0 + 16, (size_t)d * 32);
            s0 = wmma_bf16(qa[d], b0, s0);
            s1 = wmma_bf16(qa[d], b1, s1);
        }
        Frag8 S0, S1; S0.v = s0; S1.v = s1;
#pragma unroll
        for (int j = 0; j < 8; ++j) {
            float v = fmaxf(S0.f[j], S1.f[j]);
#pragma unroll
            for (int off = 1; off < 16; off <<= 1) v = fmaxf(v, __shfl_xor(v, off, 32));
            float mn = fmaxf(m[j], v);
            float p  = __expf(S0.f[j] - mn) + __expf(S1.f[j] - mn);
#pragma unroll
            for (int off = 1; off < 16; off <<= 1) p += __shfl_xor(p, off, 32);
            l[j] = l[j] * __expf(m[j] - mn) + p;
            m[j] = mn;
        }
    }
    if ((lane & 15) == 0) {
        size_t rbase = (size_t)chunk * NROW + q0 + ((lane < 16) ? 0 : 8);
#pragma unroll
        for (int j = 0; j < 8; ++j) { mpart[rbase + j] = m[j]; lpart[rbase + j] = l[j]; }
    }
}

// ---- kernel 3b: merge per-chunk (m,l) partials -----------------------------
__global__ void stats_combine_kernel(const float* __restrict__ mpart,
                                     const float* __restrict__ lpart,
                                     float* __restrict__ mrow,
                                     float* __restrict__ lrow) {
    int r = blockIdx.x * 256 + threadIdx.x;
    float m = -__builtin_inff();
#pragma unroll
    for (int c = 0; c < KCHUNKS; ++c) m = fmaxf(m, mpart[(size_t)c * NROW + r]);
    float l = 0.f;
#pragma unroll
    for (int c = 0; c < KCHUNKS; ++c)
        l += lpart[(size_t)c * NROW + r] * __expf(mpart[(size_t)c * NROW + r] - m);
    mrow[r] = m;
    lrow[r] = l;
}

// ---- kernel 4: O = (exp(S-m)/l * 1/16) @ [img|text] ------------------------
#define PSTRIDE 264  // 16x(256+8) bf16 P tile, 16B-aligned runs, conflict pad

__global__ __launch_bounds__(256) void attn_v_kernel(const __bf16* __restrict__ Q,
                                                     const __bf16* __restrict__ K,
                                                     const __bf16* __restrict__ Vt,
                                                     const float* __restrict__ mrow,
                                                     const float* __restrict__ lrow,
                                                     float* __restrict__ out) {
    __shared__ __align__(16) __bf16 Pt[16][PSTRIDE];

    int wid  = threadIdx.x >> 5;
    int lane = threadIdx.x & 31;
    int q0   = blockIdx.x * 16;

    v16bf qa[8];
#pragma unroll
    for (int d = 0; d < 8; ++d) qa[d] = load_a_bf16(Q, DIM, q0, d * 32);

    int rbase = q0 + ((lane < 16) ? 0 : 8);
    float m[8];
#pragma unroll
    for (int j = 0; j < 8; ++j) m[j] = mrow[rbase + j];

    v8f acc[4] = {};
    const int kown = wid * 32;  // this wave's key slice inside a 256-key block
    const int col0 = wid * 64;  // this wave's 64 V-columns (of 512)

    for (int kb = 0; kb < NROW; kb += 256) {
        // phase A: this wave's 16x32 score slice, exp, -> LDS (bf16)
        v8f s0 = {}, s1 = {};
#pragma unroll
        for (int d = 0; d < 8; ++d) {
            v16bf b0 = load_b_bf16(K, DIM, kb + kown,      (size_t)d * 32);
            v16bf b1 = load_b_bf16(K, DIM, kb + kown + 16, (size_t)d * 32);
            s0 = wmma_bf16(qa[d], b0, s0);
            s1 = wmma_bf16(qa[d], b1, s1);
        }
        Frag8 S0, S1; S0.v = s0; S1.v = s1;
        {
            int n  = lane & 15;
            int ro = (lane < 16) ? 0 : 8;
#pragma unroll
            for (int j = 0; j < 8; ++j) {
                Pt[ro + j][kown + n]      = (__bf16)__expf(S0.f[j] - m[j]);
                Pt[ro + j][kown + 16 + n] = (__bf16)__expf(S1.f[j] - m[j]);
            }
        }
        __syncthreads();

        // phase B: acc += P(16x256) @ Vt-slice(256 x 64) for this wave's cols
#pragma unroll
        for (int s = 0; s < 8; ++s) {
            // A-frag of P from LDS: two ds_load_b128 per lane
            int mrw = lane & 15;
            int kbA = (lane < 16) ? 0 : 8;
            const __bf16* p = &Pt[mrw][s * 32 + kbA];
            Frag16 pf;
            pf.q[0] = *reinterpret_cast<const uint4*>(p);
            pf.q[1] = *reinterpret_cast<const uint4*>(p + 16);
#pragma unroll
            for (int c = 0; c < 4; ++c) {
                v16bf bf = load_b_bf16(Vt, (size_t)NROW, col0 + c * 16,
                                       (size_t)kb + s * 32);  // B[k][n]=V[key][col]
                acc[c] = wmma_bf16(pf.v, bf, acc[c]);
            }
        }
        __syncthreads();
    }

    // epilogue: scale by (1/sqrt(256)) / l  and scatter to out_img / out_text
    float scale[8];
#pragma unroll
    for (int j = 0; j < 8; ++j) scale[j] = 0.0625f / lrow[rbase + j];
    int n  = lane & 15;
    int ro = (lane < 16) ? 0 : 8;
#pragma unroll
    for (int c = 0; c < 4; ++c) {
        Frag8 A; A.v = acc[c];
        int gcol = col0 + c * 16 + n;
        int half = (gcol < DIM) ? 0 : 1;
        float* dst = out + (size_t)half * NROW * DIM;
        int cc = gcol - half * DIM;
#pragma unroll
        for (int j = 0; j < 8; ++j)
            dst[(size_t)(q0 + ro + j) * DIM + cc] = A.f[j] * scale[j];
    }
}

// ---------------------------------------------------------------------------
extern "C" void kernel_launch(void* const* d_in, const int* in_sizes, int n_in,
                              void* d_out, int out_size, void* d_ws, size_t ws_size,
                              hipStream_t stream) {
    const float* img  = (const float*)d_in[0];
    const float* text = (const float*)d_in[1];
    const float* Wq   = (const float*)d_in[2];
    const float* Wk   = (const float*)d_in[3];

    char* ws = (char*)d_ws;
    // layout: Q bf16 (4MB) | K bf16 (4MB) | Vt bf16 (8MB) |
    //         mrow,lrow (32KB each) | mpart,lpart (256KB each)
    __bf16* Qb  = (__bf16*)(ws);
    __bf16* Kb  = (__bf16*)(ws + (size_t)4  * 1024 * 1024);
    __bf16* Vt  = (__bf16*)(ws + (size_t)8  * 1024 * 1024);
    char*   tail = ws + (size_t)16 * 1024 * 1024;
    float*  mrw = (float*)(tail);
    float*  lrw = (float*)(tail + NROW * sizeof(float));
    float*  mpt = (float*)(tail + 2 * NROW * sizeof(float));
    float*  lpt = (float*)(tail + (2 + KCHUNKS) * NROW * sizeof(float));
    float*  out = (float*)d_out;

    dim3 pgrid(NROW / 16, 4);
    proj_bf16_kernel<<<pgrid, 128, 0, stream>>>(img,  Wq, Qb);
    proj_bf16_kernel<<<pgrid, 128, 0, stream>>>(text, Wk, Kb);

    build_vt_kernel<<<dim3(NROW / 256, VW), 256, 0, stream>>>(img, text, Vt);

    attn_stats_kernel<<<dim3(NROW / 16 / 4, KCHUNKS), 128, 0, stream>>>(Qb, Kb, mpt, lpt);
    stats_combine_kernel<<<NROW / 256, 256, 0, stream>>>(mpt, lpt, mrw, lrw);

    attn_v_kernel<<<NROW / 16, 256, 0, stream>>>(Qb, Kb, Vt, mrw, lrw, out);
}
